// onlyMLP_29154238005820
// MI455X (gfx1250) — compile-verified, matrix operand order checked by reading, not run on hardware
//
#include <hip/hip_runtime.h>

typedef __bf16 bf16;
typedef unsigned int u32;
typedef __attribute__((ext_vector_type(16))) __bf16 v16bf;
typedef __attribute__((ext_vector_type(8)))  float  v8f;
typedef __attribute__((ext_vector_type(8)))  u32    v8u;
typedef __attribute__((ext_vector_type(4)))  u32    v4u;

#define N_W1 (320*256)
#define N_W2 (256*128)
#define N_W3 (128*64)

#define EPB 128           // edges per block (8 x 16-row tiles)

__device__ __forceinline__ bf16 f2bf(float f) {
    u32 u = __builtin_bit_cast(u32, f);
    u += 0x7FFFu + ((u >> 16) & 1u);                 // round-to-nearest-even
    unsigned short h = (unsigned short)(u >> 16);
    return __builtin_bit_cast(bf16, h);
}

__device__ __forceinline__ u32 pack2(float a, float b) {
    u32 lo = (u32)__builtin_bit_cast(unsigned short, f2bf(a));
    u32 hi = (u32)__builtin_bit_cast(unsigned short, f2bf(b));
    return lo | (hi << 16);
}

// 16x32 bf16 fragment (A layout; B uses the identical lane pattern over the
// out-dim because weights are stored transposed [out][in]).
// lanes 0-15 : row = base+lr, K in {kk*32+0..7,  kk*32+16..23}
// lanes 16-31: row = base+lr, K in {kk*32+8..15, kk*32+24..31}
__device__ __forceinline__ v16bf load_frag(const bf16* buf, int stride,
                                           int rcBase, int kk, int lr, int hf) {
    const bf16* p = buf + (rcBase + lr) * stride + kk * 32 + hf * 8;
    v4u a = *(const v4u*)p;
    v4u b = *(const v4u*)(p + 16);
    v8u c;
    c[0] = a[0]; c[1] = a[1]; c[2] = a[2]; c[3] = a[3];
    c[4] = b[0]; c[5] = b[1]; c[6] = b[2]; c[7] = b[3];
    return __builtin_bit_cast(v16bf, c);
}

#define WMMA_BF16(A, B, C) \
    __builtin_amdgcn_wmma_f32_16x16x32_bf16(false, (A), false, (B), (short)0, (C), false, false)

// C/D 16x16 f32 layout: VGPR vg, lane -> row = vg + 8*(lane>=16), col = lane&15
__device__ __forceinline__ void store_tile_relu(bf16* out, int stride, int rowBase,
                                                int colBase, v8f acc, float bias,
                                                int lr, int hf) {
    int col = colBase + lr;
#pragma unroll
    for (int vg = 0; vg < 8; ++vg) {
        float v = acc[vg] + bias;
        v = v > 0.0f ? v : 0.0f;
        out[(rowBase + vg + hf * 8) * stride + col] = f2bf(v);
    }
}

// ---------------- prep: convert + transpose weights to bf16 [out][in] ----------
__global__ __launch_bounds__(256) void prep_weights(
    const float* __restrict__ W1, const float* __restrict__ W2,
    const float* __restrict__ W3,
    bf16* __restrict__ W1t, bf16* __restrict__ W2t, bf16* __restrict__ W3t)
{
    int id = blockIdx.x * 256 + threadIdx.x;
    if (id < N_W1) {
        int o = id / 320, i = id % 320;
        W1t[id] = f2bf(W1[i * 256 + o]);
    } else if (id < N_W1 + N_W2) {
        int j = id - N_W1; int o = j / 256, i = j % 256;
        W2t[j] = f2bf(W2[i * 128 + o]);
    } else if (id < N_W1 + N_W2 + N_W3) {
        int j = id - (N_W1 + N_W2); int o = j / 128, i = j % 128;
        W3t[j] = f2bf(W3[i * 64 + o]);
    }
}

// ---------------- fused edge-MLP: 128 edges / block, 8 waves ----------------
__global__ __launch_bounds__(256) void edge_mlp(
    const float* __restrict__ x, const int* __restrict__ eidx,
    const float* __restrict__ ef, int E,
    const bf16* __restrict__ W1t, const float* __restrict__ b1,
    const bf16* __restrict__ W2t, const float* __restrict__ b2,
    const bf16* __restrict__ W3t, const float* __restrict__ b3,
    const float* __restrict__ W4, const float* __restrict__ b4,
    float* __restrict__ out)
{
    __shared__ bf16  sIn[EPB * 320];   // 81920 B  gathered input, bf16
    __shared__ bf16  sH1[EPB * 256];   // 65536 B
    __shared__ bf16  sH2[EPB * 128];   // 32768 B
    __shared__ float sH3[EPB * 64];    // 32768 B
    __shared__ float sPart[EPB * 2];   //  1024 B
    __shared__ int   sVU[EPB * 2];     //  1024 B   (~215 KB total; CDNA5 WGP has 320 KB)

    const int tid   = threadIdx.x;
    const int eBase = blockIdx.x * EPB;
    const int wave  = tid >> 5;
    const int lane  = tid & 31;
    const int hf    = lane >> 4;
    const int lr    = lane & 15;

    // ---- stage edge endpoint indices ----
    if (tid < EPB) {
        int e = eBase + tid;
        int ec = e < E ? e : 0;
        sVU[tid * 2 + 0] = eidx[ec];
        sVU[tid * 2 + 1] = eidx[(size_t)E + ec];
    }
    __syncthreads();

    // ---- gather [x[v] | x[u] | ef] -> bf16 LDS, float4 granularity ----
    for (int idx = tid; idx < EPB * 80; idx += 256) {
        int row = idx / 80, q = idx % 80;
        const float* src;
        if (q < 32)       src = x  + (size_t)sVU[row * 2 + 0] * 128 + q * 4;
        else if (q < 64)  src = x  + (size_t)sVU[row * 2 + 1] * 128 + (q - 32) * 4;
        else {
            int er = eBase + row; er = er < E ? er : 0;
            src = ef + (size_t)er * 64 + (q - 64) * 4;
        }
        float4 f = *(const float4*)src;
        u32* d = (u32*)&sIn[row * 320 + q * 4];
        d[0] = pack2(f.x, f.y);
        d[1] = pack2(f.z, f.w);
    }
    __syncthreads();

    // ---- layer 1: 320 -> 256  (wave: 4 row tiles x 4 col tiles) ----
    {
        const int rg = (wave & 1) * 4;       // row tile base
        const int cg = (wave >> 1) * 4;      // col tile base
        v8f acc[4][4] = {};                  // [col i][row j]
#pragma unroll 1
        for (int kk = 0; kk < 10; ++kk) {
            v16bf aF[4];
#pragma unroll
            for (int j = 0; j < 4; ++j)
                aF[j] = load_frag(sIn, 320, (rg + j) * 16, kk, lr, hf);
#pragma unroll
            for (int i = 0; i < 4; ++i) {
                v16bf bF = load_frag(W1t, 320, (cg + i) * 16, kk, lr, hf);
#pragma unroll
                for (int j = 0; j < 4; ++j)
                    acc[i][j] = WMMA_BF16(aF[j], bF, acc[i][j]);
            }
        }
#pragma unroll
        for (int i = 0; i < 4; ++i) {
            int colBase = (cg + i) * 16;
            float bias = b1[colBase + lr];
#pragma unroll
            for (int j = 0; j < 4; ++j)
                store_tile_relu(sH1, 256, (rg + j) * 16, colBase, acc[i][j], bias, lr, hf);
        }
    }
    __syncthreads();

    // ---- layer 2: 256 -> 128  (wave: 4 row tiles x 2 col tiles) ----
    {
        const int rg = (wave & 1) * 4;
        const int cg = (wave >> 1) * 2;
        v8f acc[2][4] = {};
#pragma unroll 1
        for (int kk = 0; kk < 8; ++kk) {
            v16bf aF[4];
#pragma unroll
            for (int j = 0; j < 4; ++j)
                aF[j] = load_frag(sH1, 256, (rg + j) * 16, kk, lr, hf);
#pragma unroll
            for (int i = 0; i < 2; ++i) {
                v16bf bF = load_frag(W2t, 256, (cg + i) * 16, kk, lr, hf);
#pragma unroll
                for (int j = 0; j < 4; ++j)
                    acc[i][j] = WMMA_BF16(aF[j], bF, acc[i][j]);
            }
        }
#pragma unroll
        for (int i = 0; i < 2; ++i) {
            int colBase = (cg + i) * 16;
            float bias = b2[colBase + lr];
#pragma unroll
            for (int j = 0; j < 4; ++j)
                store_tile_relu(sH2, 128, (rg + j) * 16, colBase, acc[i][j], bias, lr, hf);
        }
    }
    __syncthreads();

    // ---- layer 3: 128 -> 64  (wave: 2 row tiles x 2 col tiles), f32 out ----
    {
        const int rg = (wave & 3) * 2;
        const int cg = (wave >> 2) * 2;
        v8f acc[2][2] = {};
#pragma unroll
        for (int kk = 0; kk < 4; ++kk) {
            v16bf aF[2];
#pragma unroll
            for (int j = 0; j < 2; ++j)
                aF[j] = load_frag(sH2, 128, (rg + j) * 16, kk, lr, hf);
#pragma unroll
            for (int i = 0; i < 2; ++i) {
                v16bf bF = load_frag(W3t, 128, (cg + i) * 16, kk, lr, hf);
#pragma unroll
                for (int j = 0; j < 2; ++j)
                    acc[i][j] = WMMA_BF16(aF[j], bF, acc[i][j]);
            }
        }
#pragma unroll
        for (int i = 0; i < 2; ++i) {
            int col = (cg + i) * 16 + lr;
            float bias = b3[col];
#pragma unroll
            for (int j = 0; j < 2; ++j) {
#pragma unroll
                for (int vg = 0; vg < 8; ++vg) {
                    float v = acc[i][j][vg] + bias;
                    v = v > 0.0f ? v : 0.0f;
                    sH3[((rg + j) * 16 + vg + hf * 8) * 64 + col] = v;
                }
            }
        }
    }
    __syncthreads();

    // ---- layer 4: 64 -> 1, f32 dot + LDS reduce ----
    {
        int e = tid >> 1, p = tid & 1;          // 2 partials per edge, 32 elems each
        float s = 0.0f;
#pragma unroll
        for (int k = 0; k < 32; ++k)
            s += sH3[e * 64 + p * 32 + k] * W4[p * 32 + k];
        sPart[tid] = s;
    }
    __syncthreads();
    if (tid < EPB) {
        float s = b4[0] + sPart[tid * 2] + sPart[tid * 2 + 1];
        int e = eBase + tid;
        if (e < E) out[e] = s;
    }
}

extern "C" void kernel_launch(void* const* d_in, const int* in_sizes, int n_in,
                              void* d_out, int out_size, void* d_ws, size_t ws_size,
                              hipStream_t stream) {
    const float* x    = (const float*)d_in[0];
    const int*   eidx = (const int*)  d_in[1];
    const float* ef   = (const float*)d_in[2];
    // d_in[3] = num_nodes (unused)
    const float* W1 = (const float*)d_in[4];
    const float* b1 = (const float*)d_in[5];
    const float* W2 = (const float*)d_in[6];
    const float* b2 = (const float*)d_in[7];
    const float* W3 = (const float*)d_in[8];
    const float* b3 = (const float*)d_in[9];
    const float* W4 = (const float*)d_in[10];
    const float* b4 = (const float*)d_in[11];

    const int E = in_sizes[1] / 2;

    bf16* ws  = (bf16*)d_ws;
    bf16* W1t = ws;
    bf16* W2t = ws + N_W1;
    bf16* W3t = ws + N_W1 + N_W2;

    {
        int total = N_W1 + N_W2 + N_W3;
        int blocks = (total + 255) / 256;
        prep_weights<<<blocks, 256, 0, stream>>>(W1, W2, W3, W1t, W2t, W3t);
    }
    {
        int blocks = (E + EPB - 1) / EPB;
        edge_mlp<<<blocks, 256, 0, stream>>>(x, eidx, ef, E,
                                             W1t, b1, W2t, b2, W3t, b3, W4, b4,
                                             (float*)d_out);
    }
}
